// MultiHeadedAttention_34703335751713
// MI455X (gfx1250) — matrix operand order repaired and checked by reading, not compiled, not run on hardware
//
#include <hip/hip_runtime.h>
#include <hip/hip_bf16.h>

// ---------------------------------------------------------------------------
// MHA for MI455X (gfx1250): bf16 WMMA 16x16x32, fp32 accumulate, TDM staging.
//   B=1024, S=176 (=11*16), H=8, d=32, D=256, frames of 22 joints.
// Pass 1: Q,K = x@W + b row-major bf16; V stored transposed [d][s] bf16 in ws.
// Pass 2: per (b,h): S = QK^T/sqrt(32), temporal mask, softmax, O = P@V.
// ---------------------------------------------------------------------------

#define BATCH   1024
#define SEQ     176
#define DMODEL  256
#define HEADS   8
#define HDIM    32
#define JOINTS  22
#define NEGBIG  (-9e15f)

typedef __attribute__((ext_vector_type(16))) __bf16 v16bf;
typedef __attribute__((ext_vector_type(8)))  float  v8f;
typedef __attribute__((ext_vector_type(4)))  unsigned v4u;
typedef __attribute__((ext_vector_type(8)))  unsigned v8u;

__device__ __forceinline__ unsigned short f2bfu(float f) {
    unsigned u = __builtin_bit_cast(unsigned, f);
    return (unsigned short)((u + 0x7FFFu + ((u >> 16) & 1u)) >> 16);  // RNE
}
__device__ __forceinline__ __bf16 f2bf(float f) {
    return __builtin_bit_cast(__bf16, f2bfu(f));
}
__device__ __forceinline__ unsigned f2bf_pk(float a, float b) {
    return (unsigned)f2bfu(a) | ((unsigned)f2bfu(b) << 16);
}

struct bf16x16s { uint4 lo, hi; };

// 16 bf16 halves from LDS: halves 0..7 at p[0..7], halves 8..15 at p[gap..gap+7].
__device__ __forceinline__ v16bf ld16(const __bf16* p, int gap) {
    bf16x16s t;
    t.lo = *(const uint4*)p;
    t.hi = *(const uint4*)(p + gap);
    return __builtin_bit_cast(v16bf, t);
}

__device__ __forceinline__ v8f wmma_bf16(v16bf a, v16bf b, v8f c) {
    return __builtin_amdgcn_wmma_f32_16x16x32_bf16(
        false, a, false, b, (short)0, c, false, false);
}

// A fragment (16x32 bf16) from row-major LDS (stride in elements).
// Lane L: row=L&15, ks=(L>>4)*16; halves h -> K=ks+h  (contiguous per lane).
__device__ __forceinline__ v16bf load_a_frag(const __bf16* base, int stride, int lane) {
    return ld16(base + (lane & 15) * stride + ((lane >> 4) << 4), 8);
}
// B fragment (32x16) where the matrix is stored N-major: elem(k,n) = rows[n][k].
// Lane L: n=L&15, ks=(L>>4)*16; halves h -> K=ks+h  (contiguous per lane).
__device__ __forceinline__ v16bf load_bN_frag(const __bf16* rows, int stride, int lane) {
    return ld16(rows + (lane & 15) * stride + ((lane >> 4) << 4), 8);
}

// Tensor Data Mover: 2-D tile of bf16 (dim0 contiguous, dim1 rows) -> LDS.
// D# per CDNA5 ISA; groups 2/3 NULL (<=2D).
__device__ __forceinline__ void tdm_load_2d_bf16(const void* gsrc, void* ldst,
                                                 unsigned dim0, unsigned dim1) {
    unsigned long ga = (unsigned long)gsrc;
    unsigned la = (unsigned)(unsigned long)ldst;       // LDS byte address
    v4u g0;
    g0[0] = 1u;                                        // count=1, user descriptor
    g0[1] = la;                                        // lds_addr
    g0[2] = (unsigned)ga;                              // global_addr[31:0]
    g0[3] = (unsigned)(ga >> 32) | (2u << 30);         // global_addr[56:32], type=2
    v8u g1;
    g1[0] = 1u << 16;                                  // data_size=1 (2 bytes)
    g1[1] = dim0 << 16;                                // tensor_dim0[15:0]
    g1[2] = dim1 << 16;                                // tensor_dim1[15:0]
    g1[3] = dim0 << 16;                                // tile_dim0
    g1[4] = dim1;                                      // tile_dim1 (tile_dim2=0)
    g1[5] = dim0;                                      // tensor_dim0_stride[31:0]
    g1[6] = 0u;
    g1[7] = 0u;
    asm volatile("tensor_load_to_lds %0, %1" :: "s"(g0), "s"(g1) : "memory");
}

// ---------------------------------------------------------------------------
// Pass 1: QKV projection GEMM.  grid = 1408 (M/128), block = 256 (8 waves).
// ---------------------------------------------------------------------------
__global__ __launch_bounds__(256) void qkv_gemm_kernel(
    const float* __restrict__ x,
    const float* __restrict__ Wq, const float* __restrict__ bq,
    const float* __restrict__ Wk, const float* __restrict__ bk,
    const float* __restrict__ Wv, const float* __restrict__ bv,
    __bf16* __restrict__ ws) {

    extern __shared__ __bf16 lds[];
    __bf16* lx = lds;                 // 128 x 256 bf16 (64 KB), row-major
    __bf16* lw = lds + 128 * 256;     // W slice pre-swizzled: frag[kc][32][16] (8 KB)

    const int tid  = threadIdx.x;
    const int lane = tid & 31;
    const int wave = tid >> 5;                    // 0..7: 16-row tile owner
    const long mbase = (long)blockIdx.x * 128;

    // Stage & convert the 128x256 x-block to bf16 (2x b128 in, 1x b128 out).
    for (int i = tid * 8; i < 128 * 256; i += 256 * 8) {
        const float4 f0 = *(const float4*)&x[mbase * 256 + i];
        const float4 f1 = *(const float4*)&x[mbase * 256 + i + 4];
        uint4 p;
        p.x = f2bf_pk(f0.x, f0.y);
        p.y = f2bf_pk(f0.z, f0.w);
        p.z = f2bf_pk(f1.x, f1.y);
        p.w = f2bf_pk(f1.z, f1.w);
        *(uint4*)&lx[i] = p;
    }
    __syncthreads();

    // Hoist this wave's A fragments: invariant across all 48 output tiles.
    v16bf afrag[8];
#pragma unroll
    for (int kc = 0; kc < 8; ++kc)
        afrag[kc] = load_a_frag(lx + (wave * 16) * 256 + kc * 32, 256, lane);

    const long MATSTRIDE = (long)BATCH * HEADS * SEQ * HDIM;   // elems per Q/K/V
    const int wn   = tid & 15;          // W-stage: output column within tile
    const int kgrp = tid >> 4;          // W-stage: 16-row k group

    for (int nt = 0; nt < 48; ++nt) {
        const int sel = nt >> 4;                   // 0=Q 1=K 2=V
        const int c0  = (nt & 15) * 16;            // output column base
        const float* W  = (sel == 0) ? Wq : (sel == 1) ? Wk : Wv;
        const float* Bv = (sel == 0) ? bq : (sel == 1) ? bk : bv;

        __syncthreads();   // protect previous tile's lw reads
        // Swizzled W staging: thread covers k = kgrp*16+j, col c0+wn.
        // Dest frag[kc=kgrp>>1][L=wn+16*(kgrp&1)][h=j]: one contiguous 32B store.
        {
            unsigned wds[8];
#pragma unroll
            for (int j = 0; j < 8; ++j)
                wds[j] = f2bf_pk(W[(kgrp * 16 + 2 * j) * DMODEL + c0 + wn],
                                 W[(kgrp * 16 + 2 * j + 1) * DMODEL + c0 + wn]);
            uint4* dst = (uint4*)&lw[(((kgrp >> 1) * 32) + ((kgrp & 1) * 16) + wn) * 16];
            dst[0] = make_uint4(wds[0], wds[1], wds[2], wds[3]);
            dst[1] = make_uint4(wds[4], wds[5], wds[6], wds[7]);
        }
        if (nt + 1 < 48) {   // prefetch next weight slice (global_prefetch_b8)
            int ns  = (nt + 1) >> 4;
            int nc0 = ((nt + 1) & 15) * 16;
            const float* NW = (ns == 0) ? Wq : (ns == 1) ? Wk : Wv;
            __builtin_prefetch(&NW[(long)tid * DMODEL + nc0], 0, 1);
        }
        __syncthreads();

        v8f acc = {};
#pragma unroll
        for (int kc = 0; kc < 8; ++kc) {
            v16bf b = ld16(lw + kc * 512 + lane * 16, 8);   // pre-swizzled B frag
            acc = wmma_bf16(afrag[kc], b, acc);
        }

        // Epilogue. C layout: lane n=L&15 holds col c0+n, rows mof..mof+7.
        const int n   = lane & 15;
        const int mof = (lane >> 4) << 3;
        const float bias = Bv[c0 + n];
        const int h  = c0 >> 5;          // head
        const int d0 = c0 & 31;          // col within head
        if (sel == 2) {
            // V stored transposed [b][h][d][s]: 8 consecutive s -> one 16B store.
            long r0 = mbase + wave * 16 + mof;
            long bb = r0 / SEQ;
            int  s0 = (int)(r0 % SEQ);
            uint4 pk;
            pk.x = f2bf_pk(acc[0] + bias, acc[1] + bias);
            pk.y = f2bf_pk(acc[2] + bias, acc[3] + bias);
            pk.z = f2bf_pk(acc[4] + bias, acc[5] + bias);
            pk.w = f2bf_pk(acc[6] + bias, acc[7] + bias);
            long idx = 2 * MATSTRIDE +
                       (((bb * HEADS + h) * HDIM) + d0 + n) * SEQ + s0;
            *(uint4*)&ws[idx] = pk;
        } else {
#pragma unroll
            for (int v = 0; v < 8; ++v) {
                long r = mbase + wave * 16 + mof + v;
                long bb = r / SEQ, s = r % SEQ;
                long idx = (long)sel * MATSTRIDE +
                           (((bb * HEADS + h) * SEQ + s) * HDIM) + d0 + n;
                ws[idx] = f2bf(acc[v] + bias);
            }
        }
    }
}

// ---------------------------------------------------------------------------
// Pass 2: fused attention.  grid = 8192 (b*8+h), block = 352 (11 waves).
// Wave w owns query rows [16w, 16w+16).
// ---------------------------------------------------------------------------
__global__ __launch_bounds__(352) void attn_kernel(
    const __bf16* __restrict__ ws, float* __restrict__ out) {

    extern __shared__ __bf16 lds[];
    __bf16* lq  = lds;                        // 176x32  = 5632 (row-major [s][d])
    __bf16* lk  = lq + SEQ * HDIM;            // 176x32  = 5632 (row-major [s][d])
    __bf16* lvT = lk + SEQ * HDIM;            // 32x192  = 6144 (V^T, K padded)
    __bf16* lp  = lvT + 32 * 192;             // 176x192 = 33792 (P, K padded)

    const int tid  = threadIdx.x;
    const int lane = tid & 31;
    const int wave = tid >> 5;               // 0..10
    const int b = blockIdx.x >> 3;
    const int h = blockIdx.x & 7;

    const long MATSTRIDE = (long)BATCH * HEADS * SEQ * HDIM;
    const __bf16* qs   = ws + (long)(b * HEADS + h) * SEQ * HDIM;
    const __bf16* ksrc = qs + MATSTRIDE;
    const __bf16* vsrc = ws + 2 * MATSTRIDE + (long)(b * HEADS + h) * HDIM * SEQ;

    // Q/K tiles via Tensor Data Mover (wave 0 issues; barrier publishes).
    if (wave == 0) {
        tdm_load_2d_bf16(qs,   lq, HDIM, SEQ);
        tdm_load_2d_bf16(ksrc, lk, HDIM, SEQ);
    }
    // V^T staged manually into padded [32][192]; zero pads (uint4 granularity).
    for (int u = tid; u < 32 * 22; u += 352) {
        int d = u / 22, c = u % 22;
        ((uint4*)(lvT + d * 192))[c] = ((const uint4*)(vsrc + d * SEQ))[c];
    }
    for (int u = tid; u < 64; u += 352) {
        int d = u >> 1, c = u & 1;
        ((uint4*)(lvT + d * 192 + 176))[c] = make_uint4(0, 0, 0, 0);
    }
    {   // P pad columns (k = 176..191) zero: 176 rows x 2 uint4, 1 per thread
        int rr = tid >> 1, c = tid & 1;
        ((uint4*)(lp + rr * 192 + 176))[c] = make_uint4(0, 0, 0, 0);
    }
    if (wave == 0) __builtin_amdgcn_s_wait_tensorcnt((short)0);
    __syncthreads();

    // ---- scores: 16x176 block per wave; head_dim 32 == one WMMA K-step ----
    v8f acc[11];
    v16bf aq = load_a_frag(lq + wave * 16 * HDIM, HDIM, lane);
#pragma unroll
    for (int t = 0; t < 11; ++t) {
        v16bf bk_ = load_bN_frag(lk + t * 16 * HDIM, HDIM, lane);  // B = K^T
        v8f z = {};
        acc[t] = wmma_bf16(aq, bk_, z);
    }

    // ---- scale (folded log2(e): softmax(x) == softmax2(x*log2e)) + mask ----
    const float scale2 = 0.17677669529663687f * 1.4426950408889634f;
    const int n   = lane & 15;
    const int mof = (lane >> 4) << 3;
    int qrow[8], qfrm[8];
#pragma unroll
    for (int v = 0; v < 8; ++v) { qrow[v] = wave * 16 + mof + v; qfrm[v] = qrow[v] / JOINTS; }
#pragma unroll
    for (int t = 0; t < 11; ++t) {
        const int kp = t * 16 + n;
        const int kf = kp / JOINTS;
#pragma unroll
        for (int v = 0; v < 8; ++v) {
            float s = acc[t][v] * scale2;
            bool masked = (kf == qfrm[v]) && (kp != qrow[v]);
            acc[t][v] = masked ? NEGBIG : s;
        }
    }

    // ---- softmax over k (row m lives in one fixed 16-lane half) ----
    float rmax[8], rinv[8];
#pragma unroll
    for (int v = 0; v < 8; ++v) {
        float m = acc[0][v];
#pragma unroll
        for (int t = 1; t < 11; ++t) m = fmaxf(m, acc[t][v]);
#pragma unroll
        for (int o = 8; o >= 1; o >>= 1) m = fmaxf(m, __shfl_xor(m, o, 32));
        rmax[v] = m;
    }
#pragma unroll
    for (int v = 0; v < 8; ++v) {
        float s = 0.0f;
#pragma unroll
        for (int t = 0; t < 11; ++t) {
            float e = exp2f(acc[t][v] - rmax[v]);   // v_exp_f32, base-2 domain
            acc[t][v] = e;
            s += e;
        }
#pragma unroll
        for (int o = 8; o >= 1; o >>= 1) s += __shfl_xor(s, o, 32);
        rinv[v] = 1.0f / s;
    }

    // ---- write normalized P (bf16) to this wave's LDS rows ----
    __bf16* pw = lp + (wave * 16) * 192;
#pragma unroll
    for (int t = 0; t < 11; ++t)
#pragma unroll
        for (int v = 0; v < 8; ++v)
            pw[(mof + v) * 192 + t * 16 + n] = f2bf(acc[t][v] * rinv[v]);
    __syncthreads();

    // ---- O = P @ V : A frags hoisted over the 2 col-tiles ----
    v16bf pa[6];
#pragma unroll
    for (int kc = 0; kc < 6; ++kc)
        pa[kc] = load_a_frag(pw + kc * 32, 192, lane);
#pragma unroll
    for (int ct = 0; ct < 2; ++ct) {
        v8f o = {};
#pragma unroll
        for (int kc = 0; kc < 6; ++kc) {
            v16bf bv_ = load_bN_frag(lvT + (ct * 16) * 192 + kc * 32, 192, lane);
            o = wmma_bf16(pa[kc], bv_, o);
        }
#pragma unroll
        for (int v = 0; v < 8; ++v) {
            int q = wave * 16 + mof + v;
            out[((long)b * SEQ + q) * DMODEL + h * HDIM + ct * 16 + n] = o[v];
        }
    }
}

// ---------------------------------------------------------------------------
extern "C" void kernel_launch(void* const* d_in, const int* in_sizes, int n_in,
                              void* d_out, int out_size, void* d_ws, size_t ws_size,
                              hipStream_t stream) {
    const float* x  = (const float*)d_in[0];
    const float* Wq = (const float*)d_in[1];
    const float* bq = (const float*)d_in[2];
    const float* Wk = (const float*)d_in[3];
    const float* bk = (const float*)d_in[4];
    const float* Wv = (const float*)d_in[5];
    const float* bv = (const float*)d_in[6];
    // d_in[7] = mask: synthesized analytically in-kernel (frame structure).

    __bf16* qkv = (__bf16*)d_ws;   // 3 * 1024*8*176*32 bf16 = ~277 MB scratch
    float* out = (float*)d_out;

    const size_t lds1 = (size_t)(128 * 256 + 8 * 32 * 16) * sizeof(__bf16);   // 72 KB
    qkv_gemm_kernel<<<dim3((BATCH * SEQ) / 128), dim3(256), lds1, stream>>>(
        x, Wq, bq, Wk, bk, Wv, bv, qkv);

    const size_t lds2 = (size_t)(SEQ * HDIM * 2 + 32 * 192 + SEQ * 192) * sizeof(__bf16); // 100 KB
    attn_kernel<<<dim3(BATCH * HEADS), dim3(352), lds2, stream>>>(qkv, out);
}
